// EncoderGRU_41394894798853
// MI455X (gfx1250) — matrix operand order repaired and checked by reading, not compiled, not run on hardware
//
#include <hip/hip_runtime.h>

typedef __bf16 bf16;
typedef __attribute__((ext_vector_type(8)))  __bf16 v8bf;
typedef __attribute__((ext_vector_type(16))) __bf16 v16bf;
typedef __attribute__((ext_vector_type(8)))  float  v8f;

// Problem constants
constexpr int Bb = 128;   // batch
constexpr int Tt = 512;   // seq len
constexpr int Dd = 64;    // feature dim
constexpr int Hh = 512;   // hidden dim
constexpr int Gg = 2048;  // gate columns: [r | z | i_n | h_n]
constexpr int KH = 512;   // K from hidden state
constexpr int KX = 96;    // K from xin (65 real + pad)
constexpr int KT = 608;   // total K (19 * 32)
constexpr int KSTEPS = 19;

constexpr int NBLK = 64;
constexpr int NTHR = 256;

// Workspace layout (all 256B aligned)
constexpr size_t OFF_CTRL  = 0;                                  // 64B: ctrl[0],ctrl[1]=counters, ctrl[2]=generation
constexpr size_t OFF_AH    = 256;                                // bf16 h shadow: 128*512*2
constexpr size_t OFF_H     = OFF_AH + (size_t)Bb * KH * 2;       // f32 h: 128*512*4
constexpr size_t OFF_X     = OFF_H + (size_t)Bb * Hh * 4;        // bf16 xin padded: T*128*96*2
constexpr size_t SZ_X      = (size_t)Tt * Bb * KX * 2;
constexpr size_t OFF_WTF   = OFF_X + SZ_X;                       // teacher-forcing weights, N-major 2048x608 bf16
constexpr size_t SZ_WT     = (size_t)Gg * KT * 2;
constexpr size_t OFF_WAR   = OFF_WTF + SZ_WT;                    // autoregressive (folded) weights
constexpr size_t OFF_BTF   = OFF_WAR + SZ_WT;                    // f32 bias 2048
constexpr size_t OFF_BAR   = OFF_BTF + (size_t)Gg * 4;
constexpr size_t OFF_GATES = OFF_BAR + (size_t)Gg * 4;           // f32 gates 128x2048

__device__ inline bf16 f2bf(float f) {
  unsigned u = __float_as_uint(f);
  unsigned r = (u + 0x7FFFu + ((u >> 16) & 1u)) >> 16;
  unsigned short s = (unsigned short)r;
  return __builtin_bit_cast(bf16, s);
}

// ---------------- prep kernels ----------------

__global__ void k_init(unsigned* ctrl, unsigned* ah_words, float* h) {
  int i = blockIdx.x * blockDim.x + threadIdx.x;
  if (i < 16) ctrl[i] = 0u;
  if (i < Bb * KH / 2) ah_words[i] = 0u;   // bf16 pairs
  if (i < Bb * Hh) h[i] = 0.0f;
}

// Build padded bf16 [x, delta_t, 0...] tensor for every timestep, layout X[t][b][c], c<96
__global__ void k_prep_x(const float* __restrict__ x, const float* __restrict__ tp,
                         bf16* __restrict__ X) {
  int idx = blockIdx.x * blockDim.x + threadIdx.x;
  if (idx >= Tt * Bb * KX) return;
  int c = idx % KX;
  int b = (idx / KX) % Bb;
  int t = idx / (KX * Bb);
  float v = 0.0f;
  if (c < Dd) {
    v = x[((size_t)b * Tt + t) * Dd + c];
  } else if (c == Dd) {
    float cur = tp[(size_t)b * Tt + t];
    float prev = (t > 0) ? tp[(size_t)b * Tt + t - 1] : 0.0f;
    v = cur - prev;
  }
  X[idx] = f2bf(v);
}

// Folded weights, N-major (row n has KT contiguous K entries).
// n in [0,1024): r/z merged; [1024,1536): i_n; [1536,2048): h_n (W_hh rows 1024..1535)
__global__ void k_prep_w(const float* __restrict__ W_ih, const float* __restrict__ W_hh,
                         const float* __restrict__ W_out,
                         bf16* __restrict__ WT_tf, bf16* __restrict__ WT_ar) {
  int idx = blockIdx.x * blockDim.x + threadIdx.x;
  if (idx >= Gg * KT) return;
  int k = idx % KT;
  int n = idx / KT;
  float tf = 0.0f, ar = 0.0f;
  if (n < 1536) {
    bool is_in = (n >= 1024);  // i_n rows: no direct W_hh term
    if (k < KH) {
      float whh = is_in ? 0.0f : W_hh[(size_t)n * Hh + k];
      float dot = 0.0f;
#pragma unroll 8
      for (int d = 0; d < Dd; ++d)
        dot += W_ih[(size_t)n * (Dd + 1) + d] * W_out[(size_t)k * Dd + d];
      tf = whh;
      ar = whh + dot;          // fold y = h@W_out into recurrent weight
    } else {
      int d = k - KH;
      tf = (d < Dd + 1) ? W_ih[(size_t)n * (Dd + 1) + d] : 0.0f;
      // AR: only xin column 63 (x_t[:, -2:-1]) feeds channel 64
      ar = (d == 63) ? W_ih[(size_t)n * (Dd + 1) + Dd] : 0.0f;
    }
  } else {
    int g = n - 512;           // W_hh rows 1024..1535 -> h_n
    float w = (k < KH) ? W_hh[(size_t)g * Hh + k] : 0.0f;
    tf = w; ar = w;
  }
  WT_tf[idx] = f2bf(tf);
  WT_ar[idx] = f2bf(ar);
}

__global__ void k_prep_bias(const float* __restrict__ W_ih, const float* __restrict__ b_ih,
                            const float* __restrict__ b_hh, const float* __restrict__ b_out,
                            float* __restrict__ bias_tf, float* __restrict__ bias_ar) {
  int n = blockIdx.x * blockDim.x + threadIdx.x;
  if (n >= Gg) return;
  float btf;
  if (n < 1024)      btf = b_ih[n] + b_hh[n];
  else if (n < 1536) btf = b_ih[n];
  else               btf = b_hh[n - 512];
  float bar = btf;
  if (n < 1536) {
    float dot = 0.0f;
#pragma unroll 8
    for (int d = 0; d < Dd; ++d) dot += W_ih[(size_t)n * (Dd + 1) + d] * b_out[d];
    bar += dot;                // fold b_out@W_ih into AR bias
  }
  bias_tf[n] = btf;
  bias_ar[n] = bar;
}

// ---------------- persistent scan kernel ----------------

__device__ inline void gsync(unsigned* ctrl, unsigned& target) {
  __syncthreads();
  if (threadIdx.x == 0) {
    __builtin_amdgcn_fence(__ATOMIC_RELEASE, "agent");
    unsigned slot = target & 1u;
    target += 1u;
    unsigned prev = __hip_atomic_fetch_add(&ctrl[slot], 1u, __ATOMIC_RELAXED,
                                           __HIP_MEMORY_SCOPE_AGENT);
    if (prev == (unsigned)NBLK - 1u) {
      __hip_atomic_store(&ctrl[slot], 0u, __ATOMIC_RELAXED, __HIP_MEMORY_SCOPE_AGENT);
      __hip_atomic_fetch_add(&ctrl[2], 1u, __ATOMIC_RELEASE, __HIP_MEMORY_SCOPE_AGENT);
    } else {
      while (__hip_atomic_load(&ctrl[2], __ATOMIC_RELAXED, __HIP_MEMORY_SCOPE_AGENT) < target)
        __builtin_amdgcn_s_sleep(1);
    }
    __builtin_amdgcn_fence(__ATOMIC_ACQUIRE, "agent");
  }
  __syncthreads();
}

__global__ __launch_bounds__(NTHR) void k_scan(
    const int* __restrict__ samp_mask,
    const float* __restrict__ W_out, const float* __restrict__ b_out,
    bf16* __restrict__ A_h, float* __restrict__ h_f32,
    const bf16* __restrict__ X,
    const bf16* __restrict__ WT_tf, const bf16* __restrict__ WT_ar,
    const float* __restrict__ bias_tf, const float* __restrict__ bias_ar,
    float* __restrict__ gates, unsigned* __restrict__ ctrl,
    float* __restrict__ out) {
  const int tid  = threadIdx.x;
  const int blk  = blockIdx.x;
  const int lane = tid & 31;
  const int wave = tid >> 5;
  const int m0   = wave * 16;              // 8 waves * 16 = 128 rows
  const int mrow = m0 + (lane & 15);       // A row for this lane
  const int hi   = lane >> 4;              // half-wave selector
  const int ncol = lane & 15;
  const int n0   = blk * 32;               // 64 blocks * 32 = 2048 gate cols
  unsigned bar_target = 0;

  for (int t = 0; t < Tt; ++t) {
    const int m = samp_mask[t];
    const bf16* __restrict__ WT  = m ? WT_tf : WT_ar;
    const float* __restrict__ bs = m ? bias_tf : bias_ar;
    const bf16* __restrict__ Xt  = X + (size_t)t * (Bb * KX);

    // ---- gates GEMM: 128 x 32 slab, K = 608 ----
    v8f acc0 = {0.f, 0.f, 0.f, 0.f, 0.f, 0.f, 0.f, 0.f};
    v8f acc1 = acc0;
    const bf16* wp0 = WT + (size_t)(n0 + ncol) * KT + hi * 16;
    const bf16* wp1 = WT + (size_t)(n0 + 16 + ncol) * KT + hi * 16;
    const bf16* arow_h = A_h + (size_t)mrow * KH;
    const bf16* arow_x = Xt + (size_t)mrow * KX;
#pragma unroll
    for (int kk = 0; kk < KSTEPS; ++kk) {
      const bf16* ap = (kk < 16) ? (arow_h + kk * 32) : (arow_x + (kk - 16) * 32);
      v8bf alo = *(const v8bf*)(ap + hi * 16);
      v8bf ahi = *(const v8bf*)(ap + hi * 16 + 8);
      v16bf a = __builtin_shufflevector(alo, ahi, 0, 1, 2, 3, 4, 5, 6, 7,
                                        8, 9, 10, 11, 12, 13, 14, 15);
      v16bf b0 = *(const v16bf*)(wp0 + kk * 32);
      v16bf b1 = *(const v16bf*)(wp1 + kk * 32);
      acc0 = __builtin_amdgcn_wmma_f32_16x16x32_bf16(false, a, false, b0,
                                                     (short)0, acc0, false, false);
      acc1 = __builtin_amdgcn_wmma_f32_16x16x32_bf16(false, a, false, b1,
                                                     (short)0, acc1, false, false);
    }
    {
      const int c0 = n0 + ncol, c1 = n0 + 16 + ncol;
      const float bb0 = bs[c0], bb1 = bs[c1];
      const int rbase = m0 + (hi << 3);
#pragma unroll
      for (int v = 0; v < 8; ++v) {
        gates[(size_t)(rbase + v) * Gg + c0] = acc0[v] + bb0;
        gates[(size_t)(rbase + v) * Gg + c1] = acc1[v] + bb1;
      }
    }
    gsync(ctrl, bar_target);

    // ---- GRU elementwise: h_new, update f32 h and bf16 shadow ----
    {
      const int gtid = blk * NTHR + tid;
#pragma unroll
      for (int e = 0; e < 4; ++e) {
        int idx = e * (NBLK * NTHR) + gtid;        // 0..65535
        int b = idx >> 9, j = idx & 511;
        const float* grow = gates + (size_t)b * Gg;
        float r  = 1.0f / (1.0f + __expf(-grow[j]));
        float z  = 1.0f / (1.0f + __expf(-grow[512 + j]));
        float ng = tanhf(grow[1024 + j] + r * grow[1536 + j]);
        float hn = (1.0f - z) * ng + z * h_f32[idx];
        h_f32[idx] = hn;
        A_h[idx] = f2bf(hn);
      }
    }
    gsync(ctrl, bar_target);

    // ---- output projection y_t = h_t @ W_out + b_out (f32, overlapped with next GEMM) ----
    if (tid < 128) {
      const int row = blk * 2 + (tid >> 6);
      const int col = tid & 63;
      const float* hr = h_f32 + (size_t)row * Hh;
      float acc = b_out[col];
#pragma unroll 8
      for (int k2 = 0; k2 < Hh; ++k2) acc += hr[k2] * W_out[(size_t)k2 * Dd + col];
      out[((size_t)row * Tt + t) * Dd + col] = acc;
    }
  }
}

// ---------------- launch ----------------

extern "C" void kernel_launch(void* const* d_in, const int* in_sizes, int n_in,
                              void* d_out, int out_size, void* d_ws, size_t ws_size,
                              hipStream_t stream) {
  const float* x    = (const float*)d_in[0];
  const float* tp   = (const float*)d_in[1];
  const int* smask  = (const int*)d_in[2];
  const float* W_ih = (const float*)d_in[3];
  const float* W_hh = (const float*)d_in[4];
  const float* b_ih = (const float*)d_in[5];
  const float* b_hh = (const float*)d_in[6];
  const float* W_out = (const float*)d_in[7];
  const float* b_out = (const float*)d_in[8];
  float* out = (float*)d_out;

  char* ws = (char*)d_ws;
  unsigned* ctrl  = (unsigned*)(ws + OFF_CTRL);
  bf16* A_h       = (bf16*)(ws + OFF_AH);
  float* h_f32    = (float*)(ws + OFF_H);
  bf16* X         = (bf16*)(ws + OFF_X);
  bf16* WT_tf     = (bf16*)(ws + OFF_WTF);
  bf16* WT_ar     = (bf16*)(ws + OFF_WAR);
  float* bias_tf  = (float*)(ws + OFF_BTF);
  float* bias_ar  = (float*)(ws + OFF_BAR);
  float* gates    = (float*)(ws + OFF_GATES);

  k_init<<<dim3(256), dim3(256), 0, stream>>>(ctrl, (unsigned*)A_h, h_f32);
  k_prep_x<<<dim3((Tt * Bb * KX + 255) / 256), dim3(256), 0, stream>>>(x, tp, X);
  k_prep_w<<<dim3((Gg * KT + 255) / 256), dim3(256), 0, stream>>>(W_ih, W_hh, W_out, WT_tf, WT_ar);
  k_prep_bias<<<dim3((Gg + 255) / 256), dim3(256), 0, stream>>>(W_ih, b_ih, b_hh, b_out,
                                                                bias_tf, bias_ar);
  k_scan<<<dim3(NBLK), dim3(NTHR), 0, stream>>>(smask, W_out, b_out, A_h, h_f32, X,
                                                WT_tf, WT_ar, bias_tf, bias_ar,
                                                gates, ctrl, out);
}
